// EncoderLayer_74148315398273
// MI455X (gfx1250) — matrix operand order repaired and checked by reading, not compiled, not run on hardware
//
#include <hip/hip_runtime.h>

#define DEV_INLINE __device__ __forceinline__

typedef __attribute__((ext_vector_type(16))) __bf16 v16bf;
typedef __attribute__((ext_vector_type(8)))  float  v8f;

union FragBF { uint4 q[2]; v16bf v; };

DEV_INLINE unsigned short f2bf(float f) {
  unsigned int u = __float_as_uint(f);
  u += 0x7FFFu + ((u >> 16) & 1u);   // round-to-nearest-even
  return (unsigned short)(u >> 16);
}

// ----------------------------------------------------------------------------
// Pack fp32 W[K][Nc] (row-major) -> bf16 Wt[n][k] (transposed, contiguous K)
// ----------------------------------------------------------------------------
__global__ void pack_w_k(const float* __restrict__ W, unsigned short* __restrict__ Wt,
                         int K, int Ncol) {
  int i = blockIdx.x * 256 + threadIdx.x;
  if (i >= K * Ncol) return;
  int k = i / Ncol, n = i - k * Ncol;
  Wt[(size_t)n * K + k] = f2bf(W[i]);
}

// ----------------------------------------------------------------------------
// Generic row-tile GEMM:  C[M x Nc] = A[M x K] * Wt^T  (+bias)(+residual)(relu)
// A fp32 row-major, Wt bf16 [Nc][K].  One block = 16 rows, 8 waves loop column
// tiles.  All M here are multiples of 16.
// ----------------------------------------------------------------------------
__global__ __launch_bounds__(256) void gemm_k(
    const float* __restrict__ A, const unsigned short* __restrict__ Wt,
    const float* __restrict__ bias, const float* __restrict__ residual,
    float* __restrict__ C, int M, int Kdim, int Ncols, int relu)
{
  __shared__ unsigned short sA[16 * 272];          // K<=256, stride K+16
  const int strd = Kdim + 16;
  const int m0   = blockIdx.x * 16;
  const int tid  = threadIdx.x;

  // stage A block (fp32 -> bf16) into LDS
  const int tot = 16 * Kdim;
  for (int i = tid; i < tot; i += 256) {
    int r = i / Kdim, k = i - r * Kdim;
    int row = m0 + r;  if (row >= M) row = M - 1;
    sA[r * strd + k] = f2bf(A[(size_t)row * Kdim + k]);
  }
  __syncthreads();

  const int lane = tid & 31;
  const int wave = tid >> 5;
  const int half = lane >> 4;
  const int nc   = lane & 15;
  const int ntiles = Ncols >> 4;
  const int nk     = Kdim  >> 5;

  for (int tile = wave; tile < ntiles; tile += 8) {
    const int n0  = tile << 4;
    const int col = n0 + nc;
    v8f acc = {0.f,0.f,0.f,0.f,0.f,0.f,0.f,0.f};
    const unsigned short* wrow = Wt + (size_t)col * Kdim;
    for (int ks = 0; ks < nk; ++ks) {
      const int k0 = ks << 5;
      FragBF a, b;
      // A fragment: row = nc, K = k0+16*half .. +15 (contiguous k order, same as B)
      const uint4* ap = (const uint4*)(&sA[nc * strd + k0 + half * 16]);
      a.q[0] = ap[0];
      a.q[1] = ap[1];
      // B fragment: col = n0+nc, K = k0+16*half .. +15 (contiguous k order, same as A)
      const uint4* bp = (const uint4*)(wrow + k0 + half * 16);
      b.q[0] = bp[0];
      b.q[1] = bp[1];
      acc = __builtin_amdgcn_wmma_f32_16x16x32_bf16(false, a.v, false, b.v,
                                                    (short)0, acc, false, false);
    }
    const float bv = bias ? bias[col] : 0.f;
    for (int r = 0; r < 8; ++r) {
      const int m = m0 + r + half * 8;
      if (m < M) {
        float cv = acc[r] + bv;
        if (residual) cv += residual[(size_t)m * Ncols + col];
        if (relu)     cv  = fmaxf(cv, 0.f);
        C[(size_t)m * Ncols + col] = cv;
      }
    }
  }
}

// ----------------------------------------------------------------------------
// Fused EF = e @ We  +  attention scores  +  segment-sum scatter.
// Block = 16 edges, wave w == head w (DH == 16 == WMMA tile width).
// ----------------------------------------------------------------------------
__global__ __launch_bounds__(256) void ef_attn_k(
    const float* __restrict__ Earr, const unsigned short* __restrict__ WeT,
    const int* __restrict__ src, const int* __restrict__ dst,
    const float* __restrict__ Q, const float* __restrict__ Kmat,
    const float* __restrict__ Vmat, float* __restrict__ aV,
    float* __restrict__ zbuf)
{
  __shared__ unsigned short sE[16 * 144];          // 16 x 128 bf16, stride 144
  __shared__ int sS[16], sD[16];
  const int e0  = blockIdx.x * 16;
  const int tid = threadIdx.x;

  for (int i = tid; i < 16 * 128; i += 256) {
    int r = i >> 7, k = i & 127;
    sE[r * 144 + k] = f2bf(Earr[(size_t)(e0 + r) * 128 + k]);
  }
  if (tid < 16) { sS[tid] = src[e0 + tid]; sD[tid] = dst[e0 + tid]; }
  __syncthreads();

  const int lane = tid & 31;
  const int hh   = tid >> 5;          // head index == wave index
  const int half = lane >> 4;
  const int nc   = lane & 15;
  const int d    = hh * 16 + nc;      // feature dim within D=128

  v8f acc = {0.f,0.f,0.f,0.f,0.f,0.f,0.f,0.f};
  const unsigned short* wrow = WeT + (size_t)d * 128;
  for (int ks = 0; ks < 4; ++ks) {
    const int k0 = ks << 5;
    FragBF a, b;
    const uint4* ap = (const uint4*)(&sE[nc * 144 + k0 + half * 16]);
    a.q[0] = ap[0];
    a.q[1] = ap[1];
    const uint4* bp = (const uint4*)(wrow + k0 + half * 16);
    b.q[0] = bp[0];
    b.q[1] = bp[1];
    acc = __builtin_amdgcn_wmma_f32_16x16x32_bf16(false, a.v, false, b.v,
                                                  (short)0, acc, false, false);
  }

  // epilogue: per-edge per-head score -> exp(clamp) -> atomic segment sums
  for (int r = 0; r < 8; ++r) {
    const int m = r + half * 8;                 // edge within block
    const int s = sS[m], t = sD[m];
    float val = acc[r] * Kmat[(size_t)s * 128 + d] * Q[(size_t)t * 128 + d] * 0.25f;
    val += __shfl_xor(val, 1, 32);
    val += __shfl_xor(val, 2, 32);
    val += __shfl_xor(val, 4, 32);
    val += __shfl_xor(val, 8, 32);              // full head-dot in all 16 lanes
    float w = __expf(fminf(fmaxf(val, -5.f), 5.f));
    atomicAdd(&aV[(size_t)t * 128 + d], Vmat[(size_t)s * 128 + d] * w);
    if (nc == 0) atomicAdd(&zbuf[(size_t)t * 8 + hh], w);
  }
}

// ----------------------------------------------------------------------------
// Elementwise / reduction helpers
// ----------------------------------------------------------------------------
__global__ void zero_k(float* __restrict__ p, size_t n) {
  size_t i = (size_t)blockIdx.x * 256 + threadIdx.x;
  if (i < n) p[i] = 0.f;
}

__global__ void attn_norm_k(const float* __restrict__ aV, const float* __restrict__ z,
                            float* __restrict__ out, size_t n) {
  size_t i = (size_t)blockIdx.x * 256 + threadIdx.x;
  if (i >= n) return;
  size_t node = i >> 7;
  int h = (int)((i >> 4) & 7);
  out[i] = aV[i] / (z[node * 8 + h] + 1e-6f);
}

__global__ __launch_bounds__(128) void bn_stats_k(const float* __restrict__ X,
                                                  float* __restrict__ sum,
                                                  float* __restrict__ sq, int M) {
  int c = threadIdx.x;                 // 128 columns
  float s = 0.f, s2 = 0.f;
  for (int r = blockIdx.x; r < M; r += gridDim.x) {
    float x = X[(size_t)r * 128 + c];
    s += x; s2 += x * x;
  }
  atomicAdd(&sum[c], s);
  atomicAdd(&sq[c], s2);
}

__global__ __launch_bounds__(128) void bn_fin_k(const float* __restrict__ sum,
                                                const float* __restrict__ sq,
                                                const float* __restrict__ gamma,
                                                const float* __restrict__ beta,
                                                float* __restrict__ scale,
                                                float* __restrict__ shift, float invM) {
  int c = threadIdx.x;
  float mu  = sum[c] * invM;
  float var = sq[c] * invM - mu * mu;          // biased variance
  float sc  = rsqrtf(var + 1e-5f) * gamma[c];
  scale[c] = sc;
  shift[c] = beta[c] - mu * sc;
}

__global__ void bn_apply_k(const float* __restrict__ X, const float* __restrict__ scale,
                           const float* __restrict__ shift, float* __restrict__ Y,
                           size_t n) {
  size_t i = (size_t)blockIdx.x * 256 + threadIdx.x;
  if (i >= n) return;
  int c = (int)(i & 127);
  Y[i] = X[i] * scale[c] + shift[c];
}

__global__ void copy4_k(const float4* __restrict__ in, float4* __restrict__ out, size_t n4) {
  size_t i = (size_t)blockIdx.x * 256 + threadIdx.x;
  if (i < n4) out[i] = in[i];
}

// ----------------------------------------------------------------------------
// Host orchestration
// ----------------------------------------------------------------------------
extern "C" void kernel_launch(void* const* d_in, const int* in_sizes, int n_in,
                              void* d_out, int out_size, void* d_ws, size_t ws_size,
                              hipStream_t stream) {
  (void)in_sizes; (void)n_in; (void)out_size; (void)ws_size;
  constexpr int    Nn = 50000, Ee = 800000;
  constexpr size_t ND = (size_t)Nn * 128;       // 6,400,000
  constexpr size_t ED = (size_t)Ee * 128;       // 102,400,000

  const int*   src   = (const int*)  d_in[0];
  const int*   dst   = (const int*)  d_in[1];
  const float* h     = (const float*)d_in[2];
  const float* e     = (const float*)d_in[3];
  const float* Wq    = (const float*)d_in[5];
  const float* Wk    = (const float*)d_in[6];
  const float* Wv    = (const float*)d_in[7];
  const float* We    = (const float*)d_in[8];
  const float* Wsdp  = (const float*)d_in[9];
  const float* b_sdp = (const float*)d_in[10];
  const float* Wh1   = (const float*)d_in[11];
  const float* b_h1  = (const float*)d_in[12];
  const float* Wh2   = (const float*)d_in[13];
  const float* b_h2  = (const float*)d_in[14];
  const float* gam   = (const float*)d_in[15];
  const float* bet   = (const float*)d_in[16];

  unsigned short* wsb = (unsigned short*)d_ws;
  // bf16 weight region (ushort units)
  unsigned short* WqT   = wsb + 0;
  unsigned short* WkT   = wsb + 16384;
  unsigned short* WvT   = wsb + 32768;
  unsigned short* WeT   = wsb + 49152;
  unsigned short* WsdpT = wsb + 65536;
  unsigned short* Wh1T  = wsb + 81920;   // 256 x 128
  unsigned short* Wh2T  = wsb + 114688;  // 128 x 256
  // float region starts right after 147456 ushorts (= 73728 floats)
  float* wsf   = (float*)d_ws;
  float* SUM1  = wsf + 73728;
  float* SQ1   = SUM1 + 128;
  float* SUM2  = SQ1  + 128;
  float* SQ2   = SUM2 + 128;
  float* SCALE = SQ2  + 128;
  float* SHIFT = SCALE + 128;
  float* Qb    = wsf + 73728 + 1024;
  float* Kb    = Qb  + ND;
  float* Vb    = Kb  + ND;
  float* aV    = Vb  + ND;
  float* zb    = aV  + ND;               // N*8, contiguous after aV
  float* Tb    = zb  + (size_t)Nn * 8;   // N*256
  // reuse: h_attn->Qb, pre1->Kb, hx1->Vb, pre2->aV

  float* out_h = (float*)d_out;
  float* out_e = out_h + ND;

  // 1) pack weights to transposed bf16
  pack_w_k<<<64, 256, 0, stream>>>(Wq,   WqT,   128, 128);
  pack_w_k<<<64, 256, 0, stream>>>(Wk,   WkT,   128, 128);
  pack_w_k<<<64, 256, 0, stream>>>(Wv,   WvT,   128, 128);
  pack_w_k<<<64, 256, 0, stream>>>(We,   WeT,   128, 128);
  pack_w_k<<<64, 256, 0, stream>>>(Wsdp, WsdpT, 128, 128);
  pack_w_k<<<128, 256, 0, stream>>>(Wh1, Wh1T,  128, 256);
  pack_w_k<<<128, 256, 0, stream>>>(Wh2, Wh2T,  256, 128);

  // 2) zero accumulators (aV + z contiguous) and BN stats
  {
    size_t nz = ND + (size_t)Nn * 8;
    zero_k<<<(unsigned)((nz + 255) / 256), 256, 0, stream>>>(aV, nz);
    zero_k<<<2, 256, 0, stream>>>(SUM1, 512);
  }

  const int mblk = Nn / 16;   // 3125
  // 3) Q, K, V projections
  gemm_k<<<mblk, 256, 0, stream>>>(h, WqT, nullptr, nullptr, Qb, Nn, 128, 128, 0);
  gemm_k<<<mblk, 256, 0, stream>>>(h, WkT, nullptr, nullptr, Kb, Nn, 128, 128, 0);
  gemm_k<<<mblk, 256, 0, stream>>>(h, WvT, nullptr, nullptr, Vb, Nn, 128, 128, 0);

  // 4) fused EF GEMM + attention scatter
  ef_attn_k<<<Ee / 16, 256, 0, stream>>>(e, WeT, src, dst, Qb, Kb, Vb, aV, zb);

  // 5) h_attn = aV / (z + 1e-6)   (reuse Qb)
  attn_norm_k<<<(unsigned)((ND + 255) / 256), 256, 0, stream>>>(aV, zb, Qb, ND);

  // 6) sdp projection + residual  -> pre1 (Kb)
  gemm_k<<<mblk, 256, 0, stream>>>(Qb, WsdpT, b_sdp, h, Kb, Nn, 128, 128, 0);

  // 7) BN1 -> hx1 (Vb)
  bn_stats_k<<<1024, 128, 0, stream>>>(Kb, SUM1, SQ1, Nn);
  bn_fin_k<<<1, 128, 0, stream>>>(SUM1, SQ1, gam, bet, SCALE, SHIFT, 1.f / Nn);
  bn_apply_k<<<(unsigned)((ND + 255) / 256), 256, 0, stream>>>(Kb, SCALE, SHIFT, Vb, ND);

  // 8) FFN
  gemm_k<<<mblk, 256, 0, stream>>>(Vb, Wh1T, b_h1, nullptr, Tb, Nn, 128, 256, 1);
  gemm_k<<<mblk, 256, 0, stream>>>(Tb, Wh2T, b_h2, Vb, aV, Nn, 256, 128, 0);

  // 9) BN2 -> d_out[0 : N*D]
  bn_stats_k<<<1024, 128, 0, stream>>>(aV, SUM2, SQ2, Nn);
  bn_fin_k<<<1, 128, 0, stream>>>(SUM2, SQ2, gam, bet, SCALE, SHIFT, 1.f / Nn);
  bn_apply_k<<<(unsigned)((ND + 255) / 256), 256, 0, stream>>>(aV, SCALE, SHIFT, out_h, ND);

  // 10) e passthrough -> d_out[N*D : ]
  copy4_k<<<(unsigned)((ED / 4 + 255) / 256), 256, 0, stream>>>(
      (const float4*)e, (float4*)out_e, ED / 4);
}
